// LinearNf4Quantizer_71571335020979
// MI455X (gfx1250) — compile-verified, matrix operand order checked by reading, not run on hardware
//
#include <hip/hip_runtime.h>
#include <hip/hip_bf16.h>

typedef __bf16 bf16_t;
typedef __attribute__((ext_vector_type(4)))  __bf16 v4bf;
typedef __attribute__((ext_vector_type(8)))  __bf16 v8bf;
typedef __attribute__((ext_vector_type(16))) __bf16 v16bf;
typedef __attribute__((ext_vector_type(8)))  float  v8f;

#define OUT_F   14336
#define IN_F    4096
#define BLOCK_Q 64

#define MT 128          // block tile M
#define NT 128          // block tile N
#define KT 64           // block tile K (2 WMMA k-steps of 32)
#define LDSS 72         // padded LDS row stride in bf16 (144B -> conflict-free frag loads)

__constant__ float NF4_LUT[16] = {
    -1.0f, -0.6961928009986877f, -0.5250730514526367f, -0.39491748809814453f,
    -0.28444138169288635f, -0.18477343022823334f, -0.09105003625154495f, 0.0f,
    0.07958029955625534f, 0.16093020141124725f, 0.24611230194568634f,
    0.33791524171829224f, 0.44070982933044434f, 0.5626170039176941f,
    0.7229568362236023f, 1.0f};

// ---------------- x fp32 -> bf16 (one pass; halves repeated A traffic) ----------
__global__ void cvt_x_kernel(const float* __restrict__ x, bf16_t* __restrict__ xb,
                             int n4) {
    int i = blockIdx.x * 256 + threadIdx.x;
    if (i >= n4) return;
    const float4 v = reinterpret_cast<const float4*>(x)[i];
    v4bf o;
    o[0] = (bf16_t)v.x; o[1] = (bf16_t)v.y; o[2] = (bf16_t)v.z; o[3] = (bf16_t)v.w;
    reinterpret_cast<v4bf*>(xb)[i] = o;
}

// ---------------- bias NF4 dequant (tiny) --------------------------------------
__global__ void bias_kernel(const int* __restrict__ bp, const float* __restrict__ bam,
                            float* __restrict__ bias) {
    int i = blockIdx.x * 256 + threadIdx.x;
    if (i >= OUT_F / 2) return;
    const int b = bp[i];               // one packed byte stored per int32
    const float am = bam[i >> 5];      // 64 outputs (=32 bytes) per absmax block
    bias[2 * i]     = NF4_LUT[(b >> 4) & 15] * am;
    bias[2 * i + 1] = NF4_LUT[b & 15] * am;
}

// ---------------- fused NF4-dequant GEMM with bf16 WMMA ------------------------
__launch_bounds__(256)
__global__ void nf4_gemm_kernel(const bf16_t* __restrict__ xb,
                                const int*    __restrict__ wp,
                                const float*  __restrict__ wam,
                                const float*  __restrict__ bias,
                                float*        __restrict__ out) {
    __shared__ bf16_t As[MT * LDSS];
    __shared__ bf16_t Bs[NT * LDSS];
    __shared__ float  lut[16];

    const int tid  = threadIdx.x;
    const int lane = tid & 31;
    const int wid  = tid >> 5;     // 8 waves
    const int wm   = wid >> 2;     // 0..1 -> 64 rows each
    const int wn   = wid & 3;      // 0..3 -> 32 cols each

    const int n0 = blockIdx.x * NT;
    const int m0 = blockIdx.y * MT;

    if (tid < 16) lut[tid] = NF4_LUT[tid];

    const int row  = tid >> 1;     // 0..127 : tile row handled by this thread pair
    const int half = tid & 1;      // which 32-element half of the K tile

    const int frow  = lane & 15;   // fragment row/col within 16
    const int khalf = lane >> 4;   // which K-half this lane holds

    v8f acc[4][2];
#pragma unroll
    for (int mt = 0; mt < 4; ++mt)
#pragma unroll
        for (int nt = 0; nt < 2; ++nt) {
            v8f z = {0.f, 0.f, 0.f, 0.f, 0.f, 0.f, 0.f, 0.f};
            acc[mt][nt] = z;
        }

    __syncthreads();   // LUT visible before first dequant

    for (int k0 = 0; k0 < IN_F; k0 += KT) {
        // ---- stage A tile (bf16, already converted): 128 x 64 ----
        {
            const uint4* src = reinterpret_cast<const uint4*>(
                xb + (size_t)(m0 + row) * IN_F + k0 + half * 32);
            uint4* dst = reinterpret_cast<uint4*>(As + row * LDSS + half * 32);
            dst[0] = src[0]; dst[1] = src[1]; dst[2] = src[2]; dst[3] = src[3];
        }
        // ---- stage B tile: dequantize 128 rows of W (row n = out feature) ----
        {
            const int n = n0 + row;
            const int4* src = reinterpret_cast<const int4*>(
                wp + (size_t)n * (IN_F / 2) + (k0 >> 1) + half * 16);
            const int4 p0 = src[0], p1 = src[1], p2 = src[2], p3 = src[3];
            const float am = wam[n * (IN_F / BLOCK_Q) + (k0 >> 6)];
            const int pk[16] = {p0.x, p0.y, p0.z, p0.w, p1.x, p1.y, p1.z, p1.w,
                                p2.x, p2.y, p2.z, p2.w, p3.x, p3.y, p3.z, p3.w};
            v8bf ob[4];
#pragma unroll
            for (int q = 0; q < 4; ++q) {
#pragma unroll
                for (int e = 0; e < 4; ++e) {
                    const int b = pk[q * 4 + e];
                    ob[q][e * 2]     = (bf16_t)(lut[(b >> 4) & 15] * am); // hi nibble first
                    ob[q][e * 2 + 1] = (bf16_t)(lut[b & 15] * am);
                }
            }
            v8bf* dst = reinterpret_cast<v8bf*>(Bs + row * LDSS + half * 32);
            dst[0] = ob[0]; dst[1] = ob[1]; dst[2] = ob[2]; dst[3] = ob[3];
        }
        __syncthreads();

        // ---- two WMMA K-steps of 32 over the 64-wide tile ----
#pragma unroll
        for (int s = 0; s < 2; ++s) {
            v16bf bfrag[2];
#pragma unroll
            for (int nt = 0; nt < 2; ++nt) {
                const bf16_t* bp_ =
                    Bs + (wn * 32 + nt * 16 + frow) * LDSS + s * 32 + khalf * 8;
                const v8bf blo = *reinterpret_cast<const v8bf*>(bp_);
                const v8bf bhi = *reinterpret_cast<const v8bf*>(bp_ + 16);
                bfrag[nt] = __builtin_shufflevector(
                    blo, bhi, 0, 1, 2, 3, 4, 5, 6, 7, 8, 9, 10, 11, 12, 13, 14, 15);
            }
#pragma unroll
            for (int mt = 0; mt < 4; ++mt) {
                const bf16_t* ap_ =
                    As + (wm * 64 + mt * 16 + frow) * LDSS + s * 32 + khalf * 8;
                const v8bf alo = *reinterpret_cast<const v8bf*>(ap_);
                const v8bf ahi = *reinterpret_cast<const v8bf*>(ap_ + 16);
                const v16bf afrag = __builtin_shufflevector(
                    alo, ahi, 0, 1, 2, 3, 4, 5, 6, 7, 8, 9, 10, 11, 12, 13, 14, 15);
#pragma unroll
                for (int nt = 0; nt < 2; ++nt) {
                    acc[mt][nt] = __builtin_amdgcn_wmma_f32_16x16x32_bf16(
                        false, afrag, false, bfrag[nt], (short)0, acc[mt][nt],
                        false, false);
                }
            }
        }
        __syncthreads();
    }

    // ---- epilogue: C/D layout -> VGPR v holds M = v + 8*(lane>=16), N = lane%16
#pragma unroll
    for (int nt = 0; nt < 2; ++nt) {
        const int n = n0 + wn * 32 + nt * 16 + frow;
        const float bv = bias[n];
#pragma unroll
        for (int mt = 0; mt < 4; ++mt) {
            const int mb = m0 + wm * 64 + mt * 16 + khalf * 8;
#pragma unroll
            for (int v = 0; v < 8; ++v) {
                out[(size_t)(mb + v) * OUT_F + n] = acc[mt][nt][v] + bv;
            }
        }
    }
}

extern "C" void kernel_launch(void* const* d_in, const int* in_sizes, int n_in,
                              void* d_out, int out_size, void* d_ws, size_t ws_size,
                              hipStream_t stream) {
    const float* x        = (const float*)d_in[0];
    const int*   w_packed = (const int*)  d_in[1];
    const float* w_absmax = (const float*)d_in[2];
    const int*   b_packed = (const int*)  d_in[3];
    const float* b_absmax = (const float*)d_in[4];
    float*       out      = (float*)d_out;

    const int M = in_sizes[0] / IN_F;                 // 8192

    bf16_t* xb   = (bf16_t*)d_ws;                     // M*IN_F bf16 (64 MiB)
    float*  bias = (float*)((char*)d_ws + (size_t)M * IN_F * sizeof(bf16_t));

    const int n4 = (M * IN_F) / 4;
    cvt_x_kernel<<<(n4 + 255) / 256, 256, 0, stream>>>(x, xb, n4);
    bias_kernel<<<(OUT_F / 2 + 255) / 256, 256, 0, stream>>>(b_packed, b_absmax, bias);

    dim3 grid(OUT_F / NT, M / MT);                    // 112 x 64 blocks
    nf4_gemm_kernel<<<grid, 256, 0, stream>>>(xb, w_packed, w_absmax, bias, out);
}
